// StereoGNNKinetic_15710990368904
// MI455X (gfx1250) — compile-verified, matrix operand order checked by reading, not run on hardware
//
#include <hip/hip_runtime.h>
#include <hip/hip_bf16.h>

typedef __attribute__((ext_vector_type(16))) __bf16 v16bf;
typedef __attribute__((ext_vector_type(8)))  float  v8f;

#define NN 32768
#define EE 524288
#define BB 1024
#define HIDD 256
#define EHIDD 128

static __device__ __forceinline__ __bf16 f2bf(float f) {
  return static_cast<__bf16>(f);   // native v_cvt (RNE) on gfx1250
}

// ---------------------------------------------------------------------------
// Generic WMMA GEMM: C = act(A(M x K, lda) @ W(K x Nout, row-major LDW) + b)
// One wave per 16 x (16*NT) output tile; A fragment reused across NT WMMAs.
// Fast path: unguarded loads w/ immediate offsets. Tail: clamped-index loads
// + selects (no EXEC predication, no per-element waits).
// ---------------------------------------------------------------------------
template<int LDW, int NT>
__global__ void __launch_bounds__(32) wmma_gemm_kernel(
    const float* __restrict__ A, int lda,
    const float* __restrict__ W,
    const float* __restrict__ bias,
    float* __restrict__ C, int ldc,
    int K, int act)
{
  const int mbase  = blockIdx.x * 16;
  const int nbase0 = blockIdx.y * 16 * NT;
  const int lane   = threadIdx.x;
  const int r      = lane & 15;
  const int grp    = lane >> 4;
  v8f acc[NT];
#pragma unroll
  for (int nt = 0; nt < NT; ++nt) acc[nt] = (v8f){};
  const float* arow = A + (size_t)(mbase + r) * lda;

  int k0 = 0;
  for (; k0 + 32 <= K; k0 += 32) {
    v16bf a;
    const float* ap = arow + k0 + grp * 8;
#pragma unroll
    for (int e = 0; e < 8; ++e)  a[e] = f2bf(ap[e]);
#pragma unroll
    for (int e = 8; e < 16; ++e) a[e] = f2bf(ap[e + 8]);
    const float* wp = W + (size_t)(k0 + grp * 8) * LDW + nbase0 + r;
#pragma unroll
    for (int nt = 0; nt < NT; ++nt) {
      v16bf b;
#pragma unroll
      for (int e = 0; e < 8; ++e)  b[e] = f2bf(wp[e * LDW + nt * 16]);
#pragma unroll
      for (int e = 8; e < 16; ++e) b[e] = f2bf(wp[(e + 8) * LDW + nt * 16]);
      acc[nt] = __builtin_amdgcn_wmma_f32_16x16x32_bf16(false, a, false, b,
                                                        (short)0, acc[nt], false, false);
    }
  }
  if (k0 < K) {  // tail block: clamped loads + select, branch-free
    v16bf a;
    const int rb = k0 + grp * 8;
#pragma unroll
    for (int e = 0; e < 16; ++e) {
      int kk = rb + (e & 7) + ((e >> 3) << 4);
      int kc = (kk < K) ? kk : (K - 1);
      float av = arow[kc];
      a[e] = f2bf((kk < K) ? av : 0.0f);
    }
#pragma unroll
    for (int nt = 0; nt < NT; ++nt) {
      v16bf b;
#pragma unroll
      for (int e = 0; e < 16; ++e) {
        int kk = rb + (e & 7) + ((e >> 3) << 4);
        int kc = (kk < K) ? kk : (K - 1);
        float wv = W[(size_t)kc * LDW + nbase0 + nt * 16 + r];
        b[e] = f2bf((kk < K) ? wv : 0.0f);
      }
      acc[nt] = __builtin_amdgcn_wmma_f32_16x16x32_bf16(false, a, false, b,
                                                        (short)0, acc[nt], false, false);
    }
  }
#pragma unroll
  for (int nt = 0; nt < NT; ++nt) {
    int ncol = nbase0 + nt * 16 + r;
    float bv = bias ? bias[ncol] : 0.0f;
#pragma unroll
    for (int v = 0; v < 8; ++v) {
      float o = acc[nt][v] + bv;
      if (act == 1)      o = fmaxf(o, 0.0f);
      else if (act == 2) o = tanhf(o);
      else if (act == 3) o = 1.0f / (1.0f + __expf(-o));
      C[(size_t)(mbase + v + 8 * grp) * ldc + ncol] = o;
    }
  }
}

// ---------------------------------------------------------------------------
// Fused GAT edge-logit kernel: ze-tile = e_emb[16 edges] @ we (WMMA), then
// logits[e,head] = sum_d leaky(zl[src]+zr[dst]+ze)*att (shfl reductions).
// Avoids materializing the 536MB ze tensor (saves ~4.3GB traffic over 4 layers).
// ---------------------------------------------------------------------------
__global__ void __launch_bounds__(32) gat_edge_logits_kernel(
    const float* __restrict__ eemb,   // E x 128
    const float* __restrict__ we,     // 128 x 256
    const float* __restrict__ zl,     // N x 256
    const float* __restrict__ zr,     // N x 256
    const int* __restrict__ src, const int* __restrict__ dst,
    const float* __restrict__ att,    // 256 (H*DH)
    float* __restrict__ logits)       // E x 4
{
  const int head  = blockIdx.y;       // 0..3
  const int mbase = blockIdx.x * 16;  // edge tile
  const int lane  = threadIdx.x;
  const int r     = lane & 15;
  const int grp   = lane >> 4;
  v8f acc[4] = {{}, {}, {}, {}};
  const float* arow = eemb + (size_t)(mbase + r) * EHIDD;
#pragma unroll
  for (int k0 = 0; k0 < EHIDD; k0 += 32) {
    v16bf a;
    const float* ap = arow + k0 + grp * 8;
#pragma unroll
    for (int e = 0; e < 8; ++e)  a[e] = f2bf(ap[e]);
#pragma unroll
    for (int e = 8; e < 16; ++e) a[e] = f2bf(ap[e + 8]);
    const float* wp = we + (size_t)(k0 + grp * 8) * HIDD + head * 64 + r;
#pragma unroll
    for (int nt = 0; nt < 4; ++nt) {
      v16bf b;
#pragma unroll
      for (int e = 0; e < 8; ++e)  b[e] = f2bf(wp[e * HIDD + nt * 16]);
#pragma unroll
      for (int e = 8; e < 16; ++e) b[e] = f2bf(wp[(e + 8) * HIDD + nt * 16]);
      acc[nt] = __builtin_amdgcn_wmma_f32_16x16x32_bf16(false, a, false, b,
                                                        (short)0, acc[nt], false, false);
    }
  }
  float logit[8];
#pragma unroll
  for (int v = 0; v < 8; ++v) logit[v] = 0.0f;
#pragma unroll
  for (int nt = 0; nt < 4; ++nt) {
    int n = head * 64 + nt * 16 + r;  // this lane's feature column
    float av = att[n];
#pragma unroll
    for (int v = 0; v < 8; ++v) {
      int m = mbase + v + 8 * grp;
      int s = src[m], d = dst[m];
      float val = acc[nt][v] + zl[(size_t)s * HIDD + n] + zr[(size_t)d * HIDD + n];
      val = (val > 0.0f) ? val : 0.2f * val;   // leaky_relu 0.2
      float c = val * av;
      c += __shfl_xor(c, 1);
      c += __shfl_xor(c, 2);
      c += __shfl_xor(c, 4);
      c += __shfl_xor(c, 8);
      logit[v] += c;
    }
  }
  if (r == 0) {
#pragma unroll
    for (int v = 0; v < 8; ++v)
      logits[(size_t)(mbase + v + 8 * grp) * 4 + head] = logit[v];
  }
}

// ---------------------------------------------------------------------------
// LayerNorm: one wave per row. y = act((x+rowbias - mu)*rstd*g + b) [+ res]
// act: 0 none, 1 relu, 2 tanh, 3 exact gelu
// ---------------------------------------------------------------------------
__global__ void __launch_bounds__(32) ln_kernel(
    const float* __restrict__ X, int ldx,
    const float* __restrict__ rowbias,
    const float* __restrict__ g, const float* __restrict__ b,
    const float* __restrict__ res,
    float* __restrict__ Y, int ldy,
    int D, int act)
{
  const int row  = blockIdx.x;
  const int lane = threadIdx.x;
  float vals[8];
  float s = 0.0f;
#pragma unroll
  for (int i = 0; i < 8; ++i) {
    int c = lane + i * 32;
    float v = 0.0f;
    if (c < D) { v = X[(size_t)row * ldx + c]; if (rowbias) v += rowbias[c]; }
    vals[i] = v; s += v;
  }
#pragma unroll
  for (int m = 1; m < 32; m <<= 1) s += __shfl_xor(s, m);
  float mean = s / D;
  float sq = 0.0f;
#pragma unroll
  for (int i = 0; i < 8; ++i) {
    int c = lane + i * 32;
    float dv = (c < D) ? (vals[i] - mean) : 0.0f;
    sq += dv * dv;
  }
#pragma unroll
  for (int m = 1; m < 32; m <<= 1) sq += __shfl_xor(sq, m);
  float rstd = rsqrtf(sq / D + 1e-5f);
#pragma unroll
  for (int i = 0; i < 8; ++i) {
    int c = lane + i * 32;
    if (c >= D) continue;
    float o = (vals[i] - mean) * rstd * g[c] + b[c];
    if (act == 1)      o = fmaxf(o, 0.0f);
    else if (act == 2) o = tanhf(o);
    else if (act == 3) o = 0.5f * o * (1.0f + erff(o * 0.70710678118f));
    if (res) o += res[(size_t)row * ldy + c];
    Y[(size_t)row * ldy + c] = o;
  }
}

__global__ void fill_kernel(float* __restrict__ p, float v, size_t n) {
  size_t i = (size_t)blockIdx.x * blockDim.x + threadIdx.x;
  if (i < n) p[i] = v;
}

static __device__ __forceinline__ void atomicMaxF(float* addr, float val) {
  int* ia = (int*)addr;
  int cur = __float_as_int(*addr);
  while (__int_as_float(cur) < val) {
    int assumed = cur;
    cur = atomicCAS(ia, assumed, __float_as_int(val));
    if (cur == assumed) break;
  }
}

__global__ void seg_amax_kernel(const float* __restrict__ lg, const int* __restrict__ seg,
                                float* __restrict__ smax, size_t n) {
  size_t i = (size_t)blockIdx.x * blockDim.x + threadIdx.x;
  if (i >= n * 4) return;
  size_t e = i >> 2; int h = (int)(i & 3);
  atomicMaxF(&smax[(size_t)seg[e] * 4 + h], lg[i]);
}

__global__ void seg_expsum_kernel(const float* __restrict__ lg, const int* __restrict__ seg,
                                  const float* __restrict__ smax, float* __restrict__ ex,
                                  float* __restrict__ ssum, size_t n) {
  size_t i = (size_t)blockIdx.x * blockDim.x + threadIdx.x;
  if (i >= n * 4) return;
  size_t e = i >> 2; int h = (int)(i & 3);
  float v = __expf(lg[i] - smax[(size_t)seg[e] * 4 + h]);
  ex[i] = v;
  atomicAdd(&ssum[(size_t)seg[e] * 4 + h], v);
}

__global__ void seg_norm_kernel(float* __restrict__ ex, const int* __restrict__ seg,
                                const float* __restrict__ ssum, size_t n) {
  size_t i = (size_t)blockIdx.x * blockDim.x + threadIdx.x;
  if (i >= n * 4) return;
  size_t e = i >> 2; int h = (int)(i & 3);
  ex[i] = ex[i] / (ssum[(size_t)seg[e] * 4 + h] + 1e-16f);
}

__global__ void gat_agg_kernel(const float* __restrict__ zl, const float* __restrict__ alpha,
                               const int* __restrict__ src, const int* __restrict__ dst,
                               float* __restrict__ agg) {
  size_t i = (size_t)blockIdx.x * blockDim.x + threadIdx.x;
  if (i >= (size_t)EE * HIDD) return;
  size_t e = i >> 8; int f = (int)(i & 255);
  float a = alpha[e * 4 + (f >> 6)];
  atomicAdd(&agg[(size_t)dst[e] * HIDD + f], zl[(size_t)src[e] * HIDD + f] * a);
}

__global__ void readout_agg_kernel(const float* __restrict__ xt, const float* __restrict__ alpha,
                                   const int* __restrict__ batch, float* __restrict__ g) {
  size_t i = (size_t)blockIdx.x * blockDim.x + threadIdx.x;
  if (i >= (size_t)NN * HIDD) return;
  size_t r = i >> 8; int f = (int)(i & 255);
  atomicAdd(&g[(size_t)batch[r] * HIDD + f], alpha[r * 4 + (f >> 6)] * xt[i]);
}

__global__ void gate_kernel(const float* __restrict__ x, const float* __restrict__ gw,
                            const float* __restrict__ gb, float* __restrict__ comb) {
  int row = blockIdx.x * blockDim.x + threadIdx.x;
  if (row >= NN) return;
  const float* st = x + (size_t)row * 86 + 75;
  float s = gb[0];
  for (int k = 0; k < 11; ++k) s += st[k] * gw[k];
  float gv = 1.0f / (1.0f + __expf(-s));
  float* c = comb + (size_t)row * 288 + 256;
  for (int j = 0; j < 32; ++j) c[j] *= gv;
}

__global__ void attn_a2_kernel(const float* __restrict__ A, const float* __restrict__ W,
                               const float* __restrict__ b, float* __restrict__ out) {
  int idx = blockIdx.x * blockDim.x + threadIdx.x;
  if (idx >= NN * 4) return;
  int row = idx >> 2, o = idx & 3;
  const float* ar = A + (size_t)row * HIDD;
  float s = b[o];
  for (int k = 0; k < HIDD; ++k) s += ar[k] * W[k * 4 + o];
  out[idx] = s;
}

__global__ void head_out_kernel(const float* __restrict__ A, int K,
                                const float* __restrict__ W, const float* __restrict__ b,
                                int C, float* __restrict__ out, int ooff, int sig0) {
  int idx = blockIdx.x * blockDim.x + threadIdx.x;
  if (idx >= BB * C) return;
  int row = idx / C, c = idx % C;
  const float* ar = A + (size_t)row * K;
  float s = b[c];
  for (int k = 0; k < K; ++k) s += ar[k] * W[(size_t)k * C + c];
  if (sig0 && c == 0) s = 1.0f / (1.0f + __expf(-s));
  out[(size_t)row * 39 + ooff + c] = s;
}

// ---------------------------------------------------------------------------
// Host orchestration
// ---------------------------------------------------------------------------
static inline void gemm(hipStream_t s, const float* A, int lda, const float* W, int Nout,
                        const float* bias, float* C, int ldc, int M, int K, int act) {
  if (Nout == 256) {
    wmma_gemm_kernel<256, 4><<<dim3((unsigned)(M / 16), 4), dim3(32), 0, s>>>(A, lda, W, bias, C, ldc, K, act);
  } else if (Nout == 128) {
    wmma_gemm_kernel<128, 4><<<dim3((unsigned)(M / 16), 2), dim3(32), 0, s>>>(A, lda, W, bias, C, ldc, K, act);
  } else if (Nout == 64) {
    wmma_gemm_kernel<64, 4><<<dim3((unsigned)(M / 16), 1), dim3(32), 0, s>>>(A, lda, W, bias, C, ldc, K, act);
  } else if (Nout == 32) {
    wmma_gemm_kernel<32, 2><<<dim3((unsigned)(M / 16), 1), dim3(32), 0, s>>>(A, lda, W, bias, C, ldc, K, act);
  } else { // 16
    wmma_gemm_kernel<16, 1><<<dim3((unsigned)(M / 16), 1), dim3(32), 0, s>>>(A, lda, W, bias, C, ldc, K, act);
  }
}
static inline void lnorm(hipStream_t s, const float* X, int ldx, const float* rowbias,
                         const float* g, const float* b, const float* res,
                         float* Y, int ldy, int D, int act, int rows) {
  ln_kernel<<<dim3((unsigned)rows), dim3(32), 0, s>>>(X, ldx, rowbias, g, b, res, Y, ldy, D, act);
}
static inline void fillf(hipStream_t s, float* p, float v, size_t n) {
  fill_kernel<<<dim3((unsigned)((n + 255) / 256)), dim3(256), 0, s>>>(p, v, n);
}

extern "C" void kernel_launch(void* const* d_in, const int* in_sizes, int n_in,
                              void* d_out, int out_size, void* d_ws, size_t ws_size,
                              hipStream_t stream) {
  (void)in_sizes; (void)n_in; (void)out_size; (void)ws_size;
  const float* x     = (const float*)d_in[0];
  const int*   ei    = (const int*)d_in[1];
  const float* ea    = (const float*)d_in[2];
  const int*   batch = (const int*)d_in[3];
  const int* src = ei;
  const int* dst = ei + EE;
  float* out = (float*)d_out;
  // Param leaves in jax pytree (sorted-key) flatten order, starting at d_in[4].
  const float* const* P = (const float* const*)(d_in + 4);
  // edge_enc: 0..9 | gat[i]: 10+i*9 | kin[i]: 46+i*24 | node_enc: 118
  // readout: 136 | shared: 142 | tasks[i]: 146+i*8

  // workspace layout (floats)
  float* WS   = (float*)d_ws;
  float* h    = WS;
  float* agg  = h    + (size_t)NN * HIDD;
  float* zl   = agg  + (size_t)NN * HIDD;
  float* zr   = zl   + (size_t)NN * HIDD;
  float* comb = zr   + (size_t)NN * HIDD;     // N x 288 (also readout a1 out)
  float* smax = comb + (size_t)NN * 288;
  float* ssum = smax + (size_t)NN * 4;
  float* eemb = ssum + (size_t)NN * 4;        // E x 128
  float* ecat = eemb + (size_t)EE * EHIDD;    // E x 144 (edge concat; reused below)
  float* logits = ecat;                        // E x 4 (after edge enc done)
  float* exb    = ecat + (size_t)EE * 4;       // E x 4
  float* gbuf = ecat + (size_t)EE * 144;      // B x 256
  float* zbuf = gbuf + (size_t)BB * HIDD;     // B x 256
  float* th1  = zbuf + (size_t)BB * HIDD;     // B x 128
  float* th2  = th1  + (size_t)BB * 128;      // B x 128
  float* br   = th2  + (size_t)BB * 128;      // B x 64

  // ---------------- node encoder ----------------
  {
    const float* const* Np = P + 118;
    gemm(stream, x, 86, Np[1], HIDD, Np[0], zl, HIDD, NN, 75, 0);            // b1
    lnorm(stream, zl, HIDD, nullptr, Np[13], Np[12], nullptr, zl, HIDD, HIDD, 1, NN); // ln1+relu
    gemm(stream, zl, HIDD, Np[3], HIDD, Np[2], comb, 288, NN, HIDD, 0);      // b2 -> comb[:,0:256]
    gemm(stream, x + 75, 86, Np[15], 32, Np[14], zr, 32, NN, 11, 0);         // s1
    lnorm(stream, zr, 32, nullptr, Np[17], Np[16], nullptr, zr, 32, 32, 2, NN); // lns+tanh
    gemm(stream, zr, 32, Np[19], 32, Np[18], comb + 256, 288, NN, 32, 0);    // s2 -> comb[:,256:288]
    gate_kernel<<<dim3(NN / 256), dim3(256), 0, stream>>>(x, Np[11], Np[10], comb);
    gemm(stream, comb, 288, Np[5], HIDD, Np[4], agg, HIDD, NN, 288, 0);      // fus
    lnorm(stream, agg, HIDD, nullptr, Np[9], Np[8], nullptr, h, HIDD, HIDD, 1, NN); // lnf+relu
  }
  // ---------------- edge encoder ----------------
  {
    gemm(stream, ea, 18, P[1], EHIDD, P[0], ecat, 144, EE, 11, 0);           // b1
    lnorm(stream, ecat, 144, nullptr, P[5], P[4], nullptr, ecat, 144, EHIDD, 1, EE); // lnb+relu
    gemm(stream, ea + 11, 18, P[9], 16, P[8], ecat + 128, 144, EE, 7, 0);    // s1
    lnorm(stream, ecat + 128, 144, nullptr, P[7], P[6], nullptr, ecat + 128, 144, 16, 2, EE); // lns+tanh
    gemm(stream, ecat, 144, P[3], EHIDD, P[2], eemb, EHIDD, EE, 144, 0);     // fus
  }
  // ---------------- GAT layers ----------------
  for (int l = 0; l < 4; ++l) {
    const float* const* G = P + 10 + l * 9;
    const float* att = G[0]; const float* gbias = G[1];
    const float* ln_b = G[2]; const float* ln_g = G[3];
    const float* we = G[4];
    gemm(stream, h, HIDD, G[6], HIDD, G[5], zl, HIDD, NN, HIDD, 0);          // wl
    gemm(stream, h, HIDD, G[8], HIDD, G[7], zr, HIDD, NN, HIDD, 0);          // wr
    fillf(stream, smax, -1e30f, (size_t)NN * 4);
    fillf(stream, ssum, 0.0f, (size_t)NN * 4);
    fillf(stream, agg, 0.0f, (size_t)NN * HIDD);
    gat_edge_logits_kernel<<<dim3(EE / 16, 4), dim3(32), 0, stream>>>(
        eemb, we, zl, zr, src, dst, att, logits);
    size_t ne4 = (size_t)EE * 4;
    seg_amax_kernel<<<dim3((unsigned)((ne4 + 255) / 256)), dim3(256), 0, stream>>>(logits, dst, smax, EE);
    seg_expsum_kernel<<<dim3((unsigned)((ne4 + 255) / 256)), dim3(256), 0, stream>>>(logits, dst, smax, exb, ssum, EE);
    seg_norm_kernel<<<dim3((unsigned)((ne4 + 255) / 256)), dim3(256), 0, stream>>>(exb, dst, ssum, EE);
    gat_agg_kernel<<<dim3((unsigned)(((size_t)EE * HIDD) / 256)), dim3(256), 0, stream>>>(zl, exb, src, dst, agg);
    lnorm(stream, agg, HIDD, gbias, ln_g, ln_b, h, h, HIDD, HIDD, 1, NN);    // h = relu(ln(agg+bias)) + h
  }
  // ---------------- readout ----------------
  {
    const float* const* R = P + 136;
    gemm(stream, h, HIDD, R[1], HIDD, R[0], comb, HIDD, NN, HIDD, 2);        // tanh(a1(h))
    attn_a2_kernel<<<dim3((NN * 4) / 256), dim3(256), 0, stream>>>(comb, R[3], R[2], logits);
    fillf(stream, smax, -1e30f, (size_t)BB * 4);
    fillf(stream, ssum, 0.0f, (size_t)BB * 4);
    size_t nn4 = (size_t)NN * 4;
    seg_amax_kernel<<<dim3((unsigned)((nn4 + 255) / 256)), dim3(256), 0, stream>>>(logits, batch, smax, NN);
    seg_expsum_kernel<<<dim3((unsigned)((nn4 + 255) / 256)), dim3(256), 0, stream>>>(logits, batch, smax, exb, ssum, NN);
    seg_norm_kernel<<<dim3((unsigned)((nn4 + 255) / 256)), dim3(256), 0, stream>>>(exb, batch, ssum, NN);
    gemm(stream, h, HIDD, R[5], HIDD, R[4], zl, HIDD, NN, HIDD, 0);          // xt = tr(h)
    fillf(stream, gbuf, 0.0f, (size_t)BB * HIDD);
    readout_agg_kernel<<<dim3((unsigned)(((size_t)NN * HIDD) / 256)), dim3(256), 0, stream>>>(zl, exb, batch, gbuf);
    const float* const* S = P + 142;
    gemm(stream, gbuf, HIDD, S[1], HIDD, S[0], zbuf, HIDD, BB, HIDD, 0);     // shared.l
    lnorm(stream, zbuf, HIDD, nullptr, S[3], S[2], nullptr, zbuf, HIDD, HIDD, 1, BB); // ln+relu
  }
  // ---------------- task heads ----------------
  for (int t = 0; t < 3; ++t) {
    const float* const* T = P + 146 + t * 8;
    gemm(stream, zbuf, HIDD, T[1], 128, T[0], th1, 128, BB, HIDD, 0);        // l1
    lnorm(stream, th1, 128, nullptr, T[7], T[6], nullptr, th1, 128, 128, 1, BB); // ln1+relu
    gemm(stream, th1, 128, T[3], 64, T[2], th2, 64, BB, 128, 1);             // relu(l2)
    head_out_kernel<<<dim3((BB * 3 + 255) / 256), dim3(256), 0, stream>>>(
        th2, 64, T[5], T[4], 3, out, t * 13, 0);                             // l3
  }
  // ---------------- kinetic heads ----------------
  for (int t = 0; t < 3; ++t) {
    const float* const* Kp = P + 46 + t * 24;
    gemm(stream, zbuf, HIDD, Kp[1], 128, Kp[0], th1, 128, BB, HIDD, 0);      // e1
    lnorm(stream, th1, 128, nullptr, Kp[9], Kp[8], nullptr, th1, 128, 128, 3, BB);   // ln1+gelu
    gemm(stream, th1, 128, Kp[3], 128, Kp[2], th2, 128, BB, 128, 0);         // e2
    lnorm(stream, th2, 128, nullptr, Kp[11], Kp[10], nullptr, th2, 128, 128, 3, BB); // ln2+gelu
    gemm(stream, th2, 128, Kp[21], 64, Kp[20], br, 64, BB, 128, 1);          // pki1
    head_out_kernel<<<dim3((BB * 2 + 255) / 256), dim3(256), 0, stream>>>(br, 64, Kp[23], Kp[22], 2, out, t * 13 + 3, 0);
    gemm(stream, th2, 128, Kp[17], 64, Kp[16], br, 64, BB, 128, 1);          // pic1
    head_out_kernel<<<dim3((BB * 2 + 255) / 256), dim3(256), 0, stream>>>(br, 64, Kp[19], Kp[18], 2, out, t * 13 + 5, 0);
    gemm(stream, th2, 128, Kp[5], 64, Kp[4], br, 64, BB, 128, 1);            // kb1
    head_out_kernel<<<dim3((BB * 2 + 255) / 256), dim3(256), 0, stream>>>(br, 64, Kp[7], Kp[6], 2, out, t * 13 + 7, 1);
    gemm(stream, th2, 128, Kp[13], 64, Kp[12], br, 64, BB, 128, 1);          // md1
    head_out_kernel<<<dim3((BB * 4 + 255) / 256), dim3(256), 0, stream>>>(br, 64, Kp[15], Kp[14], 4, out, t * 13 + 9, 0);
  }
}
